// MultiHeadedAttention_RPR_17669495456455
// MI455X (gfx1250) — matrix operand order repaired and checked by reading, not compiled, not run on hardware
//
#include <hip/hip_runtime.h>
#include <hip/hip_bf16.h>
#include <stdint.h>

// ---------------------------------------------------------------------------
// MultiHeadedAttention + relative position representations for MI455X/gfx1250.
// bf16 WMMA everywhere, TDM (tensor_load_to_lds) double-buffered tile feeds,
// flash-style attention, rel-pos factored through the 33-entry vocab.
// ---------------------------------------------------------------------------

typedef __attribute__((ext_vector_type(16))) __bf16 v16bf;
typedef __attribute__((ext_vector_type(8)))  float  v8f;
typedef __attribute__((ext_vector_type(4)))  unsigned int tdm_u32x4;
typedef __attribute__((ext_vector_type(8)))  int          tdm_i32x8;
typedef __attribute__((ext_vector_type(4)))  int          tdm_i32x4;

#define B_   2
#define S_   2048
#define E_   1024
#define H_   16
#define DK_  64
#define M_   (B_ * S_)
#define RV_  33

__device__ __forceinline__ float bf2f(__bf16 x) {
  unsigned short h = __builtin_bit_cast(unsigned short, x);
  unsigned int u = ((unsigned int)h) << 16;
  return __builtin_bit_cast(float, u);
}
__device__ __forceinline__ __bf16 f2bf(float f) {
  unsigned int u = __builtin_bit_cast(unsigned int, f);
  u += 0x7fffu + ((u >> 16) & 1u);
  unsigned short h = (unsigned short)(u >> 16);
  return __builtin_bit_cast(__bf16, h);
}

// ---------------------------------------------------------------------------
// WMMA fragment loads as contiguous b128 pairs (layouts per CDNA5 ISA §7.12.2)
// A 16x32: lane half h -> runs  [kk + h*8 .. +7]  and  [kk + 16 + h*8 .. +7]
// B 32x16: lane half h -> runs  [kk + h*8 .. +7]  and  [kk + 16 + h*8 .. +7]
// ---------------------------------------------------------------------------
__device__ __forceinline__ v16bf loadA(const __bf16* tile, int ld, int row,
                                       int kk, int half) {
  union { uint4 u[2]; v16bf v; } t;
  const __bf16* p = tile + row * ld + kk + half * 8;
  t.u[0] = *(const uint4*)(p);
  t.u[1] = *(const uint4*)(p + 16);
  return t.v;
}
__device__ __forceinline__ v16bf loadB_nk(const __bf16* tile, int ld, int n,
                                          int kk, int half) {
  union { uint4 u[2]; v16bf v; } t;
  const __bf16* p = tile + n * ld + kk + half * 8;
  t.u[0] = *(const uint4*)(p);
  t.u[1] = *(const uint4*)(p + 16);
  return t.v;
}
__device__ __forceinline__ v8f wmma_bf16(v16bf a, v16bf b, v8f c) {
  return __builtin_amdgcn_wmma_f32_16x16x32_bf16(false, a, false, b,
                                                 (short)0, c, false, false);
}

// ---------------------------------------------------------------------------
// Tensor Data Mover: 2D tile (bf16) global -> LDS.  D# per 08_async_tensor §8.
// width/height/stride in bf16 elements; descriptor is wave-uniform.
// ---------------------------------------------------------------------------
__device__ __forceinline__ void tdm_load_2d(unsigned lds_byte_addr,
                                            const void* gsrc,
                                            unsigned width, unsigned height,
                                            unsigned stride) {
  unsigned long long ga = (unsigned long long)(uintptr_t)gsrc;
  tdm_u32x4 g0;
  g0.x = 1u;                                                 // count=1 (valid)
  g0.y = lds_byte_addr;                                      // lds_addr
  g0.z = (unsigned)ga;                                       // global_addr lo
  g0.w = (unsigned)((ga >> 32) & 0x01ffffffu) | (2u << 30);  // ga hi | type=2
  tdm_i32x8 g1;
  g1[0] = (int)(1u << 16);                                   // data_size=1 (2B)
  g1[1] = (int)((width & 0xffffu) << 16);                    // tensor_dim0 lo16
  g1[2] = (int)(((width >> 16) & 0xffffu) |
                ((height & 0xffffu) << 16));                 // dim0 hi | dim1 lo
  g1[3] = (int)(((height >> 16) & 0xffffu) |
                ((width & 0xffffu) << 16));                  // dim1 hi | tile_dim0
  g1[4] = (int)(height & 0xffffu);                           // tile_dim1 (tile_dim2=0)
  g1[5] = (int)stride;                                       // tensor_dim0_stride lo
  g1[6] = 0;
  g1[7] = 0;
  tdm_i32x4 z4 = (tdm_i32x4)0;
#if __clang_major__ >= 23
  tdm_i32x8 z8 = (tdm_i32x8)0;
  __builtin_amdgcn_tensor_load_to_lds(g0, g1, z4, z4, z8, 0);
#else
  __builtin_amdgcn_tensor_load_to_lds(g0, g1, z4, z4, 0);
#endif
}
__device__ __forceinline__ unsigned lds_addr_of(const void* p) {
  return (unsigned)(uintptr_t)p;  // low 32 bits of generic LDS address = offset
}

// ---------------------------------------------------------------------------
__global__ void cast_bf16_kernel(const float* __restrict__ src,
                                 __bf16* __restrict__ dst, int n) {
  int i = blockIdx.x * 256 + threadIdx.x;
  if (i < n) dst[i] = f2bf(src[i]);
}

// ---------------------------------------------------------------------------
// Y = X @ W^T + bias.  MODE 0: f32 [M,N]; MODE 1: bf16 [B,H,S,DK];
// MODE 2: bf16 [B,H,DK,S] (transposed, for V).  TDM double-buffered tiles.
// ---------------------------------------------------------------------------
template <int MODE>
__global__ __launch_bounds__(128)
void gemm_bf16_kernel(const __bf16* __restrict__ A,
                      const __bf16* __restrict__ W,
                      const float* __restrict__ bias,
                      void* __restrict__ out, int M, int N, int K) {
  __shared__ __align__(16) __bf16 At[2][64 * 64];
  __shared__ __align__(16) __bf16 Wt[2][64 * 64];

  const int tid = threadIdx.x;
  const int wave = tid >> 5, lane = tid & 31;
  const int half = lane >> 4, ln = lane & 15;
  const int m_block = blockIdx.y * 64;
  const int n_block = blockIdx.x * 64;

  v8f acc[4];
#pragma unroll
  for (int t = 0; t < 4; ++t)
#pragma unroll
    for (int r = 0; r < 8; ++r) acc[t][r] = 0.f;

  const int nk = K / 64;
  if (wave == 0) {
    tdm_load_2d(lds_addr_of(At[0]), A + (size_t)m_block * K, 64, 64, K);
    tdm_load_2d(lds_addr_of(Wt[0]), W + (size_t)n_block * K, 64, 64, K);
  }
  for (int kb = 0; kb < nk; ++kb) {
    const int cur = kb & 1;
    if (wave == 0) {
      if (kb + 1 < nk) {
        tdm_load_2d(lds_addr_of(At[cur ^ 1]),
                    A + (size_t)m_block * K + (kb + 1) * 64, 64, 64, K);
        tdm_load_2d(lds_addr_of(Wt[cur ^ 1]),
                    W + (size_t)n_block * K + (kb + 1) * 64, 64, 64, K);
        __builtin_amdgcn_s_wait_tensorcnt(2);
      } else {
        __builtin_amdgcn_s_wait_tensorcnt(0);
      }
    }
    __syncthreads();
#pragma unroll
    for (int kk = 0; kk < 2; ++kk) {
      v16bf a = loadA(At[cur], 64, 16 * wave + ln, kk * 32, half);
#pragma unroll
      for (int t = 0; t < 4; ++t) {
        v16bf b = loadB_nk(Wt[cur], 64, 16 * t + ln, kk * 32, half);
        acc[t] = wmma_bf16(a, b, acc[t]);
      }
    }
    __syncthreads();
  }

#pragma unroll
  for (int t = 0; t < 4; ++t) {
    int n_g = n_block + 16 * t + ln;
    float bv = bias[n_g];
    if (MODE == 2) {
      // 8 consecutive s-rows -> one uint4 store into [B,H,DK,S]
      int h = n_g >> 6, d = n_g & 63;
      int m0 = m_block + 16 * wave + 8 * half;
      int b = m0 / S_, s0 = m0 % S_;
      union { __bf16 e[8]; uint4 u; } pk;
#pragma unroll
      for (int r = 0; r < 8; ++r) pk.e[r] = f2bf(acc[t][r] + bv);
      *(uint4*)((__bf16*)out + (((size_t)b * H_ + h) * DK_ + d) * S_ + s0) = pk.u;
    } else {
#pragma unroll
      for (int r = 0; r < 8; ++r) {
        int m_g = m_block + 16 * wave + r + 8 * half;
        float val = acc[t][r] + bv;
        if (MODE == 1) {
          int h = n_g >> 6, d = n_g & 63;
          int b = m_g / S_, s = m_g % S_;
          ((__bf16*)out)[(((size_t)b * H_ + h) * S_ + s) * DK_ + d] = f2bf(val);
        } else {
          ((float*)out)[(size_t)m_g * N + n_g] = val;
        }
      }
    }
  }
}

// ---------------------------------------------------------------------------
// Qr[bh*S+s][r] = sum_d Q[bh,s,d] * rel_k_table[r,d]
// ---------------------------------------------------------------------------
__global__ void qr_kernel(const __bf16* __restrict__ Qp,
                          const float* __restrict__ relk,
                          float* __restrict__ Qr, int total) {
  int i = blockIdx.x * 256 + threadIdx.x;
  if (i >= total) return;
  int r = i % RV_;
  int qs = i / RV_;
  const __bf16* q = Qp + (size_t)qs * DK_;
  const float* t = relk + r * DK_;
  float acc = 0.f;
#pragma unroll
  for (int d = 0; d < DK_; ++d) acc += bf2f(q[d]) * t[d];
  Qr[i] = acc;
}

// ---------------------------------------------------------------------------
// Flash attention. Grid (S/64, B*H), 128 thr = 4 waves, 16 q-rows per wave.
// K tiles [s][d] and V tiles [d][s] (from transposed Vp) streamed by TDM with
// double buffering; Qr bias staged in LDS; rel-v via 33 bucket sums (ds_add_f32).
// ---------------------------------------------------------------------------
__global__ __launch_bounds__(128)
void attn_kernel(const __bf16* __restrict__ Qp, const __bf16* __restrict__ Kp,
                 const __bf16* __restrict__ VpT, const float* __restrict__ Qr,
                 const __bf16* __restrict__ relv, __bf16* __restrict__ ctx) {
  __shared__ union ShMem {
    struct { __bf16 Kt[2][64 * 64]; __bf16 Vt[2][64 * 64]; } s;  // loop phase
    float Ot[64 * 64];                                           // epilogue
  } sh;
  __shared__ __align__(16) __bf16 Pt[4 * 16 * 64];  // Q staging, then P tiles
  __shared__ float buckets[64][RV_];
  __shared__ float Qrt[64][RV_];
  __shared__ float rowScale[64];
  __shared__ float rowL[64];

  const int bh = blockIdx.y;
  const int q0 = blockIdx.x * 64;
  const int tid = threadIdx.x;
  const int wave = tid >> 5, lane = tid & 31;
  const int half = lane >> 4, ln = lane & 15;

  const __bf16* Qb = Qp + (size_t)bh * S_ * DK_;
  const __bf16* Kb = Kp + (size_t)bh * S_ * DK_;
  const __bf16* Vb = VpT + (size_t)bh * DK_ * S_;   // [d][s]
  const float* Qrb = Qr + (size_t)bh * S_ * RV_;

  // Q tile -> Pt via TDM; meanwhile stage Qr rows + zero buckets
  if (wave == 0) tdm_load_2d(lds_addr_of(Pt), Qb + (size_t)q0 * DK_, 64, 64, 64);
  for (int idx = tid; idx < 64 * RV_; idx += 128) {
    int row = idx / RV_, r = idx % RV_;
    buckets[row][r] = 0.f;
    Qrt[row][r] = Qrb[(size_t)(q0 + row) * RV_ + r];
  }
  if (wave == 0) __builtin_amdgcn_s_wait_tensorcnt(0);
  __syncthreads();
  v16bf aq[2];
  aq[0] = loadA(Pt, 64, 16 * wave + ln, 0, half);
  aq[1] = loadA(Pt, 64, 16 * wave + ln, 32, half);
  __syncthreads();

  v8f o[4];
  float mrow[8], lrow[8];
#pragma unroll
  for (int t = 0; t < 4; ++t)
#pragma unroll
    for (int r = 0; r < 8; ++r) o[t][r] = 0.f;
#pragma unroll
  for (int r = 0; r < 8; ++r) { mrow[r] = -1e30f; lrow[r] = 0.f; }

  const int nkb = S_ / 64;
  if (wave == 0) {
    tdm_load_2d(lds_addr_of(sh.s.Kt[0]), Kb, 64, 64, 64);
    tdm_load_2d(lds_addr_of(sh.s.Vt[0]), Vb, 64, 64, S_);
  }
  for (int kb = 0; kb < nkb; ++kb) {
    const int cur = kb & 1;
    const int k0 = kb * 64;
    if (wave == 0) {
      if (kb + 1 < nkb) {
        tdm_load_2d(lds_addr_of(sh.s.Kt[cur ^ 1]),
                    Kb + (size_t)(k0 + 64) * DK_, 64, 64, 64);
        tdm_load_2d(lds_addr_of(sh.s.Vt[cur ^ 1]),
                    Vb + (k0 + 64), 64, 64, S_);
        __builtin_amdgcn_s_wait_tensorcnt(2);
      } else {
        __builtin_amdgcn_s_wait_tensorcnt(0);
      }
    }
    __syncthreads();  // tiles for kb ready; prior readers of buf cur are done

    // logits S = Q K^T
    v8f sa[4];
#pragma unroll
    for (int t = 0; t < 4; ++t)
#pragma unroll
      for (int r = 0; r < 8; ++r) sa[t][r] = 0.f;
#pragma unroll
    for (int kk = 0; kk < 2; ++kk) {
#pragma unroll
      for (int t = 0; t < 4; ++t) {
        v16bf b = loadB_nk(sh.s.Kt[cur], 64, 16 * t + ln, kk * 32, half);
        sa[t] = wmma_bf16(aq[kk], b, sa[t]);
      }
    }

    // + rel-k bias from LDS, row max
    float sv[4][8], mx[8];
#pragma unroll
    for (int r = 0; r < 8; ++r) mx[r] = -1e30f;
#pragma unroll
    for (int t = 0; t < 4; ++t) {
#pragma unroll
      for (int r = 0; r < 8; ++r) {
        int qrow = 16 * wave + r + 8 * half;
        int dlt = (k0 + 16 * t + ln) - (q0 + qrow);
        if (dlt > 16) dlt = 16;
        if (dlt < -16) dlt = -16;
        float val = sa[t][r] + Qrt[qrow][dlt + 16];
        sv[t][r] = val;
        mx[r] = fmaxf(mx[r], val);
      }
    }
#pragma unroll
    for (int m = 1; m < 16; m <<= 1)
#pragma unroll
      for (int r = 0; r < 8; ++r) mx[r] = fmaxf(mx[r], __shfl_xor(mx[r], m, 32));

    float sc[8], rs[8];
#pragma unroll
    for (int r = 0; r < 8; ++r) {
      float mnew = fmaxf(mrow[r], mx[r]);
      sc[r] = __expf(mrow[r] - mnew);
      mrow[r] = mnew;
      rs[r] = 0.f;
    }
#pragma unroll
    for (int t = 0; t < 4; ++t)
#pragma unroll
      for (int r = 0; r < 8; ++r) {
        float p = __expf(sv[t][r] - mrow[r]);
        sv[t][r] = p;
        rs[r] += p;
      }
#pragma unroll
    for (int m = 1; m < 16; m <<= 1)
#pragma unroll
      for (int r = 0; r < 8; ++r) rs[r] += __shfl_xor(rs[r], m, 32);
#pragma unroll
    for (int r = 0; r < 8; ++r) lrow[r] = lrow[r] * sc[r] + rs[r];
#pragma unroll
    for (int t = 0; t < 4; ++t)
#pragma unroll
      for (int r = 0; r < 8; ++r) o[t][r] *= sc[r];

    if (ln == 0)
#pragma unroll
      for (int r = 0; r < 8; ++r) rowScale[16 * wave + r + 8 * half] = sc[r];
    __syncthreads();
    for (int idx = tid; idx < 64 * RV_; idx += 128)
      buckets[idx / RV_][idx % RV_] *= rowScale[idx / RV_];
    __syncthreads();

    // bucket accumulation (ds_add_f32) + P staging (A layout: row-contiguous)
#pragma unroll
    for (int t = 0; t < 4; ++t) {
#pragma unroll
      for (int r = 0; r < 8; ++r) {
        int qrow = 16 * wave + r + 8 * half;
        int dlt = (k0 + 16 * t + ln) - (q0 + qrow);
        if (dlt > 16) dlt = 16;
        if (dlt < -16) dlt = -16;
        atomicAdd(&buckets[qrow][dlt + 16], sv[t][r]);
        Pt[wave * 1024 + (r + 8 * half) * 64 + 16 * t + ln] = f2bf(sv[t][r]);
      }
    }

    // O += P V   (Vt holds [d][k] so B-fragments are contiguous)
#pragma unroll
    for (int kk = 0; kk < 2; ++kk) {
      v16bf ap = loadA(Pt + wave * 1024, 64, ln, kk * 32, half);
#pragma unroll
      for (int t = 0; t < 4; ++t) {
        v16bf b = loadB_nk(sh.s.Vt[cur], 64, 16 * t + ln, kk * 32, half);
        o[t] = wmma_bf16(ap, b, o[t]);
      }
    }
    __syncthreads();  // all reads of buf cur done before next TDM overwrite
  }

  // epilogue: normalize, add bucket x rel_v, write context
  if (ln == 0)
#pragma unroll
    for (int r = 0; r < 8; ++r) rowL[16 * wave + r + 8 * half] = lrow[r];
#pragma unroll
  for (int t = 0; t < 4; ++t)
#pragma unroll
    for (int r = 0; r < 8; ++r)
      sh.Ot[(16 * wave + r + 8 * half) * 64 + 16 * t + ln] = o[t][r] / lrow[r];
  __syncthreads();

  const int b = bh / H_, h = bh % H_;
  for (int idx = tid; idx < 64 * 64; idx += 128) {
    int row = idx >> 6, d = idx & 63;
    float acc = sh.Ot[row * 64 + d];
    float linv = 1.0f / rowL[row];
#pragma unroll
    for (int rr = 0; rr < RV_; ++rr)
      acc += buckets[row][rr] * linv * bf2f(relv[rr * DK_ + d]);
    int sg = q0 + row;
    ctx[((size_t)b * S_ + sg) * E_ + h * DK_ + d] = f2bf(acc);
  }
}

// ---------------------------------------------------------------------------
extern "C" void kernel_launch(void* const* d_in, const int* in_sizes, int n_in,
                              void* d_out, int out_size, void* d_ws,
                              size_t ws_size, hipStream_t stream) {
  const float* query = (const float*)d_in[0];
  const float* key_  = (const float*)d_in[1];
  const float* value = (const float*)d_in[2];
  const float* Wq = (const float*)d_in[3];  const float* bq = (const float*)d_in[4];
  const float* Wk = (const float*)d_in[5];  const float* bk = (const float*)d_in[6];
  const float* Wv = (const float*)d_in[7];  const float* bv = (const float*)d_in[8];
  const float* Wo = (const float*)d_in[9];  const float* bo = (const float*)d_in[10];
  const float* relk = (const float*)d_in[11];
  const float* relv = (const float*)d_in[12];
  float* out = (float*)d_out;

  char* ws = (char*)d_ws;
  size_t off = 0;
  auto carve = [&](size_t bytes) -> void* {
    void* p = ws + off;
    off = (off + bytes + 255) & ~(size_t)255;
    return p;
  };
  const size_t actB = (size_t)M_ * E_ * sizeof(__bf16);
  const size_t wB   = (size_t)E_ * E_ * sizeof(__bf16);
  __bf16* xq_bf = (__bf16*)carve(actB);
  __bf16* xk_bf = (__bf16*)carve(actB);
  __bf16* xv_bf = (__bf16*)carve(actB);
  __bf16* wq_bf = (__bf16*)carve(wB);
  __bf16* wk_bf = (__bf16*)carve(wB);
  __bf16* wv_bf = (__bf16*)carve(wB);
  __bf16* wo_bf = (__bf16*)carve(wB);
  __bf16* relv_bf = (__bf16*)carve((size_t)RV_ * DK_ * sizeof(__bf16));
  __bf16* Qp  = (__bf16*)carve(actB);
  __bf16* Kp  = (__bf16*)carve(actB);
  __bf16* VpT = (__bf16*)carve(actB);
  float*  Qr  = (float*)carve((size_t)B_ * H_ * S_ * RV_ * sizeof(float));
  __bf16* ctx = (__bf16*)carve(actB);

  auto cast = [&](const float* s, __bf16* d, int n) {
    cast_bf16_kernel<<<(n + 255) / 256, 256, 0, stream>>>(s, d, n);
  };
  cast(query, xq_bf, M_ * E_);
  cast(key_,  xk_bf, M_ * E_);
  cast(value, xv_bf, M_ * E_);
  cast(Wq, wq_bf, E_ * E_);
  cast(Wk, wk_bf, E_ * E_);
  cast(Wv, wv_bf, E_ * E_);
  cast(Wo, wo_bf, E_ * E_);
  cast(relv, relv_bf, RV_ * DK_);

  dim3 ggrid(E_ / 64, M_ / 64);
  gemm_bf16_kernel<1><<<ggrid, 128, 0, stream>>>(xq_bf, wq_bf, bq, Qp, M_, E_, E_);
  gemm_bf16_kernel<1><<<ggrid, 128, 0, stream>>>(xk_bf, wk_bf, bk, Kp, M_, E_, E_);
  gemm_bf16_kernel<2><<<ggrid, 128, 0, stream>>>(xv_bf, wv_bf, bv, VpT, M_, E_, E_);

  int qr_total = B_ * H_ * S_ * RV_;
  qr_kernel<<<(qr_total + 255) / 256, 256, 0, stream>>>(Qp, relk, Qr, qr_total);

  dim3 agrid(S_ / 64, B_ * H_);
  attn_kernel<<<agrid, 128, 0, stream>>>(Qp, Kp, VpT, Qr, relv_bf, ctx);

  gemm_bf16_kernel<0><<<ggrid, 128, 0, stream>>>(ctx, wo_bf, bo, out, M_, E_, E_);
}
